// NeuralMemory_9646496546825
// MI455X (gfx1250) — compile-verified, matrix operand order checked by reading, not run on hardware
//
#include <hip/hip_runtime.h>
#include <hip/hip_bf16.h>

typedef unsigned short u16;
typedef __attribute__((ext_vector_type(16))) __bf16 v16bf;
typedef __attribute__((ext_vector_type(8)))  __bf16 v8bf;
typedef __attribute__((ext_vector_type(8)))  float  v8f;

#define WAVES_PER_BLOCK 8

// ---------- bf16 helpers (explicit RNE, bit-cast; no fptrunc dependence) ----------
__device__ inline u16 f2bfbits(float f) {
    unsigned u = __float_as_uint(f);
    unsigned r = u + 0x7FFFu + ((u >> 16) & 1u);
    return (u16)(r >> 16);
}
__device__ inline float bf2f(u16 b) {
    return __uint_as_float(((unsigned)b) << 16);
}
__device__ inline __bf16 bits2bf(u16 b) {
    return __builtin_bit_cast(__bf16, b);
}

// Contiguous-K fragment: base points at (row/col)*stride + k0; lane's 16 elements are
// two 16-byte chunks per ISA 7.12.2 (K = khalf+0..7 and 16+khalf+0..7).
__device__ inline v16bf frag_contig(const u16* base, int khalf) {
    v8bf lo = *(const v8bf*)(base + khalf);
    v8bf hi = *(const v8bf*)(base + 16 + khalf);
    return __builtin_shufflevector(lo, hi, 0, 1, 2, 3, 4, 5, 6, 7,
                                           8, 9, 10, 11, 12, 13, 14, 15);
}

// ---------- one 16x16 output tile, unrolled K-loop of v_wmma_f32_16x16x32_bf16 ----------
// A logical (16 x K): element (m,k) at Ab[m*sam + k*sak]
// B logical (K x 16): element (k,n) at Bb[k*sbk + n*sbn]
// AC: sak==1 (K-contiguous A).  BC: sbk==1 (K-contiguous B).
template <int K, bool AC, bool BC>
__device__ inline v8f tile_mma(const u16* Ab, int sam, int sak,
                               const u16* Bb, int sbk, int sbn,
                               int lane, v8f acc) {
    const int m     = lane & 15;           // row of A / col of B for this lane
    const int khalf = (lane >> 4) << 3;    // 0 or 8
#pragma unroll
    for (int k0 = 0; k0 < K; k0 += 32) {
        v16bf a, b;
        if (AC) {
            a = frag_contig(Ab + m * sam + k0, khalf);
        } else {
#pragma unroll
            for (int e = 0; e < 16; ++e) {
                int kk = k0 + ((e < 8) ? (khalf + e) : (8 + khalf + e));
                a[e] = bits2bf(Ab[m * sam + kk * sak]);
            }
        }
        if (BC) {
            b = frag_contig(Bb + m * sbn + k0, khalf);
        } else {
#pragma unroll
            for (int e = 0; e < 16; ++e) {
                int kk = k0 + ((e < 8) ? (khalf + e) : (8 + khalf + e));
                b[e] = bits2bf(Bb[kk * sbk + m * sbn]);
            }
        }
        acc = __builtin_amdgcn_wmma_f32_16x16x32_bf16(
            /*neg_a=*/false, a, /*neg_b=*/false, b,
            /*c_mod=*/(short)0, acc, /*reuse_a=*/false, /*reuse_b=*/false);
    }
    return acc;
}

// ---------- tiled GEMM over a (Mt x Nt) grid of 16x16 tiles, waves round-robin ----------
template <int Mt, int Nt, int K, bool AC, bool BC, typename Put>
__device__ inline void gemm_tiles(const u16* A, int sam, int sak,
                                  const u16* B, int sbk, int sbn,
                                  int wid, int lane, Put put) {
    for (int ti = wid; ti < Mt * Nt; ti += WAVES_PER_BLOCK) {
        int tm = ti / Nt;
        int tn = ti - tm * Nt;
        v8f acc = {};
        acc = tile_mma<K, AC, BC>(A + tm * 16 * sam, sam, sak,
                                  B + tn * 16 * sbn, sbk, sbn, lane, acc);
#pragma unroll
        for (int r = 0; r < 8; ++r) {
            // ISA 7.12.2 C/D layout: m = r + 8*(lane>=16), n = lane&15
            int gm = tm * 16 + r + ((lane >> 4) << 3);
            int gn = tn * 16 + (lane & 15);
            put(gm, gn, acc[r]);
        }
    }
}

// ================= kernel 1: convert weights fp32 -> bf16, pre-transposed =================
// wqT/wkT/wv1T/wkvT: stored [out][in] so B-side fragments are K-contiguous.
// wv2 kept in both layouts (used as wv2 and wv2^T).
__global__ void prep_kernel(const float* w_kv, const float* wq, const float* wk,
                            const float* wv1, const float* wv2,
                            u16* wkvT, u16* wqT, u16* wkT, u16* wv1T,
                            u16* wv2b, u16* wv2bT) {
    int i = blockIdx.x * 256 + threadIdx.x;
    if (i < 512 * 1024) {
        int k = i >> 10, n = i & 1023;           // w_kv row-major [512][1024]
        wkvT[n * 512 + k] = f2bfbits(w_kv[i]);
    }
    int j = i - 512 * 1024;
    if (j >= 0 && j < 4 * 16384) {
        int m = j >> 14, r = j & 16383;
        int k = r >> 7, n = r & 127;             // [128][128] row-major
        const float* src = (m == 0) ? wq : (m == 1) ? wk : (m == 2) ? wv1 : wv2;
        u16 v = f2bfbits(src[r]);
        if (m == 0)      wqT[n * 128 + k]  = v;
        else if (m == 1) wkT[n * 128 + k]  = v;
        else if (m == 2) wv1T[n * 128 + k] = v;
        else { wv2b[r] = v; wv2bT[n * 128 + k] = v; }
    }
}

// ================= kernel 2: RMSNorm, one block per row =================
__global__ void rmsnorm_kernel(const float* __restrict__ seq, const float* __restrict__ nw,
                               float* __restrict__ xf, u16* __restrict__ xbw) {
    __shared__ float red[256];
    const int row = blockIdx.x, t = threadIdx.x;
    const float* src = seq + (size_t)row * 512;
    float a0 = src[t], a1 = src[t + 256];
    red[t] = a0 * a0 + a1 * a1;
    __syncthreads();
    for (int s = 128; s > 0; s >>= 1) {
        if (t < s) red[t] += red[t + s];
        __syncthreads();
    }
    float scale = rsqrtf(red[0] * (1.0f / 512.0f) + 1.1920928955078125e-07f);
    float o0 = a0 * scale * nw[t];
    float o1 = a1 * scale * nw[t + 256];
    size_t base = (size_t)row * 512;
    xf[base + t] = o0;            xf[base + t + 256] = o1;
    xbw[base + t] = f2bfbits(o0); xbw[base + t + 256] = f2bfbits(o1);
}

// ================= kernel 3: per-chunk control signals =================
__global__ void control_kernel(const float* __restrict__ xf,
                               const float* __restrict__ w_step,
                               const float* __restrict__ w_mom,
                               const float* __restrict__ w_decay,
                               float* lr, float* mo, float* de) {
    __shared__ float xcs[512];
    const int b = blockIdx.x >> 5;   // 0..1
    const int n = blockIdx.x & 31;   // 0..31
    const int t = threadIdx.x;
    for (int d = t; d < 512; d += 256) {
        float s = 0.0f;
        const float* base = xf + ((size_t)(b * 2048 + n * 64)) * 512 + d;
        for (int c = 0; c < 64; ++c) s += base[(size_t)c * 512];
        xcs[d] = s * (1.0f / 64.0f);
    }
    __syncthreads();
    if (t < 12) {
        int mt = t >> 2, h = t & 3;
        const float* W = (mt == 0) ? w_step : (mt == 1) ? w_mom : w_decay;
        float acc = 0.0f;
        for (int d = 0; d < 512; ++d) acc += xcs[d] * W[d * 4 + h];
        float sig = 1.0f / (1.0f + __expf(-acc));
        int idx = (b * 4 + h) * 32 + n;
        if (mt == 0)      lr[idx] = __expf(sig * -15.0f);   // adaptive_lr
        else if (mt == 1) mo[idx] = sig;                    // adaptive_mom
        else              de[idx] = 1.0f - sig;             // (1 - decay) gate
    }
}

// ================= kernel 4: kv = x @ w_kv (both operands K-contiguous) =================
__global__ void kv_gemm_kernel(const u16* __restrict__ xbw, const u16* __restrict__ wkvT,
                               u16* __restrict__ keysb, u16* __restrict__ valsb) {
    const int wid = threadIdx.x >> 5, lane = threadIdx.x & 31;
    const int tile = blockIdx.x * WAVES_PER_BLOCK + wid;   // 0..16383
    const int tm = tile >> 6;   // row tile 0..255  (M = 4096)
    const int tn = tile & 63;   // col tile 0..63   (N = 1024)
    if (lane == 0) __builtin_prefetch(wkvT + (size_t)tn * 16 * 512, 0, 0);
    v8f acc = {};
    acc = tile_mma<512, true, true>(xbw + (size_t)tm * 16 * 512, 512, 1,
                                    wkvT + (size_t)tn * 16 * 512, 1, 512, lane, acc);
#pragma unroll
    for (int r = 0; r < 8; ++r) {
        int m = tm * 16 + r + ((lane >> 4) << 3);   // global row in [0,4096)
        int n = tn * 16 + (lane & 15);              // global col in [0,1024)
        int b = m >> 11, s = m & 2047, cn = s >> 6, ci = s & 63;
        int c = n & 511, h = c >> 7, d = c & 127;
        size_t idx = (((size_t)((b * 4 + h) * 32 + cn)) * 64 + ci) * 128 + d;
        u16 v = f2bfbits(acc[r]);
        if (n < 512) keysb[idx] = v;
        else         valsb[idx] = v;
    }
}

// ================= kernel 5: per-chunk autograd of the memory model =================
// One block (8 waves) per chunk; all intermediates LDS-resident (304 KB).
// Dual-layout copies make nearly every WMMA operand K-contiguous.
__global__ void grad_kernel(const u16* __restrict__ keysb, const u16* __restrict__ valsb,
                            const u16* __restrict__ wqT, const u16* __restrict__ wkT,
                            const u16* __restrict__ wv1T, const u16* __restrict__ wv2b,
                            const u16* __restrict__ wv2bT,
                            const float* __restrict__ lr, float* __restrict__ out) {
    extern __shared__ char smem_raw[];
    u16* xb   = (u16*)smem_raw;   // x   [row][dim]
    u16* xbT  = xb   + 8192;      // x^T [dim][row]
    u16* tb   = xbT  + 8192;      // target
    u16* qb   = tb   + 8192;      // q   [row][dim]
    u16* kb   = qb   + 8192;      // k   [row][dim]
    u16* kbT  = kb   + 8192;      // k^T [dim][row]
    u16* vb   = kbT  + 8192;      // v   [row][dim]
    u16* vbT  = vb   + 8192;      // v^T [dim][row]
    u16* sb   = vbT  + 8192;      // silu(hidden)      [row][dim]
    u16* sbT  = sb   + 8192;      // silu^T            [dim][row]
    u16* db   = sbT  + 8192;      // silu'(hidden)
    u16* dpb  = db   + 8192;      // dpred [row][dim]   -> reused as dq^T [dim][row]
    u16* dpT  = dpb  + 8192;      // dpred^T
    u16* dhb  = dpT  + 8192;      // dh [row][dim]      -> reused as dk^T [dim][row]
    u16* dhT  = dhb  + 8192;      // dh^T
    u16* dvT  = dhT  + 8192;      // dv^T [dim][row] (only layout needed)
    u16* Pb   = dvT  + 8192;      // probs bf16 (64x64)
    u16* dSb  = Pb   + 4096;      // scaled softmax-backward (64x64)
    float* Pf = (float*)(dSb + 4096); // probs f32 (64x64)
    float* Df = Pf + 4096;            // dP f32    (64x64)

    const int chunk = blockIdx.x;            // = bh*32 + n
    const int tid = threadIdx.x, wid = tid >> 5, lane = tid & 31;
    const float nlr = -lr[chunk];
    float* ob = out + (size_t)chunk * 16384; // + f*FS for grad index f
    const size_t FS = (size_t)256 * 16384;
    const float scl = 0.08838834764831845f;  // 128^-0.5

    if (tid == 0) {
        __builtin_prefetch(wqT, 0, 0);  __builtin_prefetch(wkT, 0, 0);
        __builtin_prefetch(wv1T, 0, 0); __builtin_prefetch(wv2bT, 0, 0);
    }

    // P0: stage x (both layouts) / target into LDS
    {
        const uint4* xg4 = (const uint4*)(keysb + (size_t)chunk * 8192);
        const uint4* tg4 = (const uint4*)(valsb + (size_t)chunk * 8192);
        uint4* xb4 = (uint4*)xb;
        uint4* tb4 = (uint4*)tb;
        for (int i = tid; i < 1024; i += 256) {
            uint4 xv = xg4[i], tv = tg4[i];
            xb4[i] = xv; tb4[i] = tv;
            int r = i >> 4;             // row (16 uint4 per 128-elem row)
            int d0 = (i & 15) * 8;      // first dim of this chunk
            const u16* p = (const u16*)&xv;
#pragma unroll
            for (int e = 0; e < 8; ++e) xbT[(d0 + e) * 64 + r] = p[e];
        }
    }
    __syncthreads();

    // P1: q = x@wq, k = x@wk, v = x@wv1   (B = pre-transposed weights, K-contiguous)
    gemm_tiles<4, 8, 128, true, true>(xb, 128, 1, wqT, 1, 128, wid, lane,
               [&](int m, int n, float v) { qb[m * 128 + n] = f2bfbits(v); });
    gemm_tiles<4, 8, 128, true, true>(xb, 128, 1, wkT, 1, 128, wid, lane,
               [&](int m, int n, float v) { u16 x = f2bfbits(v); kb[m * 128 + n] = x; kbT[n * 64 + m] = x; });
    gemm_tiles<4, 8, 128, true, true>(xb, 128, 1, wv1T, 1, 128, wid, lane,
               [&](int m, int n, float v) { u16 x = f2bfbits(v); vb[m * 128 + n] = x; vbT[n * 64 + m] = x; });
    __syncthreads();

    // P2: scores = scale * q @ k^T   (B = k natural is K-contiguous for k^T)
    gemm_tiles<4, 4, 128, true, true>(qb, 128, 1, kb, 1, 128, wid, lane,
               [&](int m, int n, float v) { Pf[m * 64 + n] = v * scl; });
    __syncthreads();

    // P3: causal softmax, row per thread
    if (tid < 64) {
        int i = tid;
        float mx = -3.0e38f;
        for (int j = 0; j <= i; ++j) mx = fmaxf(mx, Pf[i * 64 + j]);
        float sum = 0.0f;
        for (int j = 0; j <= i; ++j) { float e = __expf(Pf[i * 64 + j] - mx); Pf[i * 64 + j] = e; sum += e; }
        float inv = 1.0f / sum;
        for (int j = 0; j < 64; ++j) {
            float p = (j <= i) ? Pf[i * 64 + j] * inv : 0.0f;
            Pf[i * 64 + j] = p;
            Pb[i * 64 + j] = f2bfbits(p);
        }
    }
    __syncthreads();

    // P4: hidden = P @ v (B = v^T layout, K-contiguous); fuse silu + silu'
    gemm_tiles<4, 8, 64, true, true>(Pb, 64, 1, vbT, 1, 64, wid, lane,
               [&](int m, int n, float h) {
                   float sg = 1.0f / (1.0f + __expf(-h));
                   u16 s = f2bfbits(h * sg);
                   sb[m * 128 + n] = s; sbT[n * 64 + m] = s;
                   db[m * 128 + n] = f2bfbits(sg * (1.0f + h * (1.0f - sg)));
               });
    __syncthreads();

    // P5: pred = silu @ wv2 (B = wv2^T layout); fuse dpred = 2*(pred-target)/DH
    gemm_tiles<4, 8, 128, true, true>(sb, 128, 1, wv2bT, 1, 128, wid, lane,
               [&](int m, int n, float p) {
                   u16 x = f2bfbits((p - bf2f(tb[m * 128 + n])) * (2.0f / 128.0f));
                   dpb[m * 128 + n] = x; dpT[n * 64 + m] = x;
               });
    __syncthreads();

    // P6: gwv2 = silu^T @ dpred (f=3);  dhidden = (dpred @ wv2^T) * silu'
    gemm_tiles<8, 8, 64, true, true>(sbT, 64, 1, dpT, 1, 64, wid, lane,
               [&](int m, int n, float g) { ob[3 * FS + (size_t)m * 128 + n] = g * nlr; });
    gemm_tiles<4, 8, 128, true, true>(dpb, 128, 1, wv2b, 1, 128, wid, lane,
               [&](int m, int n, float v) {
                   u16 x = f2bfbits(v * bf2f(db[m * 128 + n]));
                   dhb[m * 128 + n] = x; dhT[n * 64 + m] = x;
               });
    __syncthreads();

    // P8: dP = dh @ v^T (B = v natural, K-contiguous);  dv = P^T @ dh (store dv^T only)
    gemm_tiles<4, 4, 128, true, true>(dhb, 128, 1, vb, 1, 128, wid, lane,
               [&](int m, int n, float v) { Df[m * 64 + n] = v; });
    gemm_tiles<4, 8, 64, false, true>(Pb, 1, 64, dhT, 1, 64, wid, lane,
               [&](int m, int n, float v) { dvT[n * 64 + m] = f2bfbits(v); });
    __syncthreads();

    // P9: softmax backward: dS = scale * P * (dP - rowsum(dP*P))
    if (tid < 64) {
        int i = tid;
        float rd = 0.0f;
        for (int j = 0; j <= i; ++j) rd += Df[i * 64 + j] * Pf[i * 64 + j];
        for (int j = 0; j < 64; ++j)
            dSb[i * 64 + j] = f2bfbits(Pf[i * 64 + j] * (Df[i * 64 + j] - rd) * scl);
    }
    __syncthreads();

    // P10: gwv1 = x^T @ dv (f=2);  dq = dS @ k (store dq^T);  dk = dS^T @ q (store dk^T)
    gemm_tiles<8, 8, 64, true, true>(xbT, 64, 1, dvT, 1, 64, wid, lane,
               [&](int m, int n, float g) { ob[2 * FS + (size_t)m * 128 + n] = g * nlr; });
    gemm_tiles<4, 8, 64, true, true>(dSb, 64, 1, kbT, 1, 64, wid, lane,
               [&](int m, int n, float v) { dpb[n * 64 + m] = f2bfbits(v); }); // dq^T in dpb slot
    gemm_tiles<4, 8, 64, false, false>(dSb, 1, 64, qb, 128, 1, wid, lane,
               [&](int m, int n, float v) { dhb[n * 64 + m] = f2bfbits(v); }); // dk^T in dhb slot
    __syncthreads();

    // P11: gwq = x^T @ dq (f=0); gwk = x^T @ dk (f=1)  (all operands K-contiguous)
    gemm_tiles<8, 8, 64, true, true>(xbT, 64, 1, dpb, 1, 64, wid, lane,
               [&](int m, int n, float g) { ob[0 * FS + (size_t)m * 128 + n] = g * nlr; });
    gemm_tiles<8, 8, 64, true, true>(xbT, 64, 1, dhb, 1, 64, wid, lane,
               [&](int m, int n, float g) { ob[1 * FS + (size_t)m * 128 + n] = g * nlr; });
}

// ================= kernel 6: fused momentum + decay scans (in-place on d_out) =================
__global__ void scan_kernel(const float* __restrict__ mo, const float* __restrict__ de,
                            float* __restrict__ out) {
    const int tid = blockIdx.x * 256 + threadIdx.x;   // 524288 threads
    const int f   = tid >> 17;         // / (8*16384)
    const int rem = tid & 131071;
    const int bh  = rem >> 14;
    const int e   = rem & 16383;
    size_t base = ((size_t)(f * 256) + (size_t)bh * 32) * 16384 + e;
    float m = 0.0f, u = 0.0f;
    for (int n = 0; n < 32; ++n) {
        float g1 = mo[bh * 32 + n];
        float g2 = de[bh * 32 + n];
        float s = out[base + (size_t)n * 16384];
        m = s + g1 * m;   // momentum scan (h_0 = x_0)
        u = m + g2 * u;   // decay scan    (u_0 = m_0)
        out[base + (size_t)n * 16384] = u;
    }
}

// ================= host launcher =================
extern "C" void kernel_launch(void* const* d_in, const int* in_sizes, int n_in,
                              void* d_out, int out_size, void* d_ws, size_t ws_size,
                              hipStream_t stream) {
    (void)in_sizes; (void)n_in; (void)out_size; (void)ws_size;
    const float* seq     = (const float*)d_in[0];
    const float* norm_w  = (const float*)d_in[1];
    const float* wq      = (const float*)d_in[2];
    const float* wk      = (const float*)d_in[3];
    const float* wv1     = (const float*)d_in[4];
    const float* wv2     = (const float*)d_in[5];
    const float* w_kv    = (const float*)d_in[6];
    const float* w_step  = (const float*)d_in[7];
    const float* w_mom   = (const float*)d_in[8];
    const float* w_decay = (const float*)d_in[9];
    float* out = (float*)d_out;

    char* ws = (char*)d_ws;
    float* xf    = (float*)ws;                        // 4096*512 f32   = 8 MiB
    u16*   xbw   = (u16*)(ws + 8388608);              // 4096*512 bf16  = 4 MiB
    u16*   wkvT  = (u16*)(ws + 12582912);             // 1024*512 bf16  = 1 MiB
    u16*   wqT   = (u16*)(ws + 13631488);             // 5 x 16384 bf16
    u16*   wkT   = wqT   + 16384;
    u16*   wv1T  = wkT   + 16384;
    u16*   wv2b  = wv1T  + 16384;
    u16*   wv2bT = wv2b  + 16384;
    u16*   keysb = (u16*)(ws + 13795328);             // 256*64*128 bf16 = 4 MiB
    u16*   valsb = keysb + 2097152;                   // 4 MiB
    float* lr = (float*)(ws + 22183936);              // 256 f32 each
    float* mo = lr + 256;
    float* de = mo + 256;

    prep_kernel<<<2304, 256, 0, stream>>>(w_kv, wq, wk, wv1, wv2,
                                          wkvT, wqT, wkT, wv1T, wv2b, wv2bT);
    rmsnorm_kernel<<<4096, 256, 0, stream>>>(seq, norm_w, xf, xbw);
    control_kernel<<<64, 256, 0, stream>>>(xf, w_step, w_mom, w_decay, lr, mo, de);
    kv_gemm_kernel<<<2048, 256, 0, stream>>>(xbw, wkvT, keysb, valsb);

    const size_t smem = (16 * 8192 + 2 * 4096) * sizeof(u16) + 2 * 4096 * sizeof(float); // 311296 B
    grad_kernel<<<256, 256, smem, stream>>>(keysb, valsb, wqT, wkT, wv1T, wv2b, wv2bT, lr, out);

    scan_kernel<<<2048, 256, 0, stream>>>(mo, de, out);
}
